// CustomRobertaEncoder_2551210574127
// MI455X (gfx1250) — compile-verified, matrix operand order checked by reading, not run on hardware
//
#include <hip/hip_runtime.h>
#include <math.h>

// ---------------------------------------------------------------------------
// CDNA5 WMMA (wave32): A/B = 16 bf16 per lane, C/D = 8 f32 per lane.
// Per ISA 7.12.2, a lane's 16 A elements are two contiguous K-runs of 8:
//   k in [8*half, 8*half+8) and [16+8*half, 16+8*half+8), half = lane>>4.
// Storing tiles K-contiguous per row makes every fragment 2x ds_load_b128.
// B fragments use the mirrored mapping with n = lane&15, so the B tile is
// kept N-major ([n][k], i.e. transposed) in LDS/global.
// Tile staging uses gfx1250 async global->LDS DMA (ASYNCcnt-tracked).
// ---------------------------------------------------------------------------
typedef unsigned int       u32;
typedef unsigned short     u16;
typedef __attribute__((ext_vector_type(16))) __bf16 v16bf;
typedef __attribute__((ext_vector_type(8)))  float  v8f;

__device__ __forceinline__ u16 f32_to_bf16(float f) {
  u32 u = __float_as_uint(f);
  u32 r = u + 0x7FFFu + ((u >> 16) & 1u);   // round-to-nearest-even
  return (u16)(r >> 16);
}

__device__ __forceinline__ v8f zero8() {
  v8f z = {0.f, 0.f, 0.f, 0.f, 0.f, 0.f, 0.f, 0.f};
  return z;
}

// Async DMA: 16 bytes global -> LDS, no VGPR round-trip (ISA 15.18.3 op 98).
// LDS dest VGPR takes the LDS byte address = low 32 bits of the generic
// shared-space pointer (ISA 10.2: LDS aperture uses addr[31:0]).
__device__ __forceinline__ void stage_async128(u16* ldst, const u16* gsrc) {
  asm volatile("global_load_async_to_lds_b128 %0, %1, off"
               :: "v"((u32)(size_t)ldst), "v"(gsrc)
               : "memory");
}

__device__ __forceinline__ void wait_async0() {
#if __has_builtin(__builtin_amdgcn_s_wait_asynccnt)
  __builtin_amdgcn_s_wait_asynccnt(0);
#else
  asm volatile("s_wait_asynccnt 0x0" ::: "memory");
#endif
}

// Fragment load from a K-contiguous LDS tile: row = row0 + (lane&15).
// Two b128 loads cover the lane's two 8-element K-runs.
__device__ __forceinline__ v16bf load_frag16(const u16* lds, int row0, int k0,
                                             int stride, int lane) {
  const int row = row0 + (lane & 15), half = lane >> 4;
  const u16* base = lds + (size_t)row * stride + k0 + 8 * half;
  union { uint4 q[2]; v16bf v; } r;
  r.q[0] = *(const uint4*)(base);        // k = k0+8h .. +7
  r.q[1] = *(const uint4*)(base + 16);   // k = k0+16+8h .. +7
  return r.v;
}

__device__ __forceinline__ v8f wmma_bf16(v16bf a, v16bf b, v8f c) {
  return __builtin_amdgcn_wmma_f32_16x16x32_bf16(
      /*neg_a=*/false, a, /*neg_b=*/false, b,
      /*c_mod=*/(short)0, c, /*reuse_a=*/false, /*reuse_b=*/false);
}

// ---------------------------------------------------------------------------
// fp32 -> bf16 cast (row-major copy)
// ---------------------------------------------------------------------------
__global__ void cast_bf16(const float* __restrict__ in, u16* __restrict__ out,
                          int n) {
  for (int i = blockIdx.x * blockDim.x + threadIdx.x; i < n;
       i += gridDim.x * blockDim.x)
    out[i] = f32_to_bf16(in[i]);
}

// fp32 [K][N] -> bf16 [N][K] transposing cast (32x32 tiles via LDS).
__global__ __launch_bounds__(256) void cast_transpose_bf16(
    const float* __restrict__ in, u16* __restrict__ out, int K, int N) {
  __shared__ float tile[32][33];
  const int k0 = blockIdx.y * 32, n0 = blockIdx.x * 32;
  const int tx = threadIdx.x & 31, ty = threadIdx.x >> 5;  // 32 x 8
#pragma unroll
  for (int i = 0; i < 4; ++i)
    tile[ty + i * 8][tx] = in[(size_t)(k0 + ty + i * 8) * N + n0 + tx];
  __syncthreads();
#pragma unroll
  for (int i = 0; i < 4; ++i)
    out[(size_t)(n0 + ty + i * 8) * K + k0 + tx] =
        f32_to_bf16(tile[tx][ty + i * 8]);
}

// ---------------------------------------------------------------------------
// bf16 WMMA GEMM: C[M,N] = A[M,K] * B[K,N] + bias (B given transposed [N][K]).
// 256 threads (8 wave32), tile 128x128, BK=64 (two WMMA k-steps per stage).
// Wave grid 4(M) x 2(N); each wave owns 32x64 = 8 accumulators.
// ---------------------------------------------------------------------------
#define MODE_F32  0   // outF row-major f32
#define MODE_BF16 1   // outB row-major bf16
#define MODE_GELU 2   // gelu(x) -> outB row-major bf16
#define MODE_QV   3   // scatter to [B,NH,S,HD] bf16 (Q and K layout)
#define MODE_KT   4   // scatter to [B,NH,HD,S] bf16 (V layout)

__global__ __launch_bounds__(256) void gemm_bf16(
    const u16* __restrict__ A, const u16* __restrict__ Bt,
    const float* __restrict__ bias, float* __restrict__ outF,
    u16* __restrict__ outB, int M, int N, int K, int mode) {
  __shared__ __align__(16) u16 As[128 * 64];  // 16 KB, [m][k]
  __shared__ __align__(16) u16 Bs[128 * 64];  // 16 KB, [n][k]

  const int tid = threadIdx.x, lane = tid & 31, wave = tid >> 5;
  const int wm = wave & 3, wn = wave >> 2;
  const int m0 = blockIdx.y * 128, n0 = blockIdx.x * 128;

  v8f acc[2][4];
#pragma unroll
  for (int i = 0; i < 2; ++i)
#pragma unroll
    for (int j = 0; j < 4; ++j) acc[i][j] = zero8();

  const int sRow = tid >> 1, sCol = (tid & 1) * 32;  // 2 thr/row, 64B each

  for (int kb = 0; kb < K; kb += 64) {
    {  // async-stage A[m][k] and Bt[n][k] tiles (128x64 each), 8x16B per lane
      const u16* ga = A + (size_t)(m0 + sRow) * K + kb + sCol;
      const u16* gb = Bt + (size_t)(n0 + sRow) * K + kb + sCol;
      u16* la = As + sRow * 64 + sCol;
      u16* lb = Bs + sRow * 64 + sCol;
#pragma unroll
      for (int c = 0; c < 4; ++c) {
        stage_async128(la + c * 8, ga + c * 8);
        stage_async128(lb + c * 8, gb + c * 8);
      }
    }
    wait_async0();      // this wave's DMAs landed in LDS
    __syncthreads();    // all waves' tiles visible

#pragma unroll
    for (int ks = 0; ks < 2; ++ks) {
      const v16bf a0 = load_frag16(As, wm * 32, ks * 32, 64, lane);
      const v16bf a1 = load_frag16(As, wm * 32 + 16, ks * 32, 64, lane);
#pragma unroll
      for (int nf = 0; nf < 4; ++nf) {
        const v16bf b = load_frag16(Bs, wn * 64 + nf * 16, ks * 32, 64, lane);
        acc[0][nf] = wmma_bf16(a0, b, acc[0][nf]);
        acc[1][nf] = wmma_bf16(a1, b, acc[1][nf]);
      }
    }
    __syncthreads();    // nobody still reads the tile before next DMA wave
  }

  // Epilogue. C/D layout: row = r + 8*half, col = lane&15 within a 16x16 frag.
  const int half = lane >> 4, ln = lane & 15;
#pragma unroll
  for (int mi = 0; mi < 2; ++mi) {
#pragma unroll
    for (int nf = 0; nf < 4; ++nf) {
      const int n = n0 + wn * 64 + nf * 16 + ln;
      const float bv = bias[n];
#pragma unroll
      for (int r = 0; r < 8; ++r) {
        const int m = m0 + wm * 32 + mi * 16 + r + half * 8;
        float v = acc[mi][nf][r] + bv;
        switch (mode) {
          case MODE_F32:
            outF[(size_t)m * N + n] = v;
            break;
          case MODE_BF16:
            outB[(size_t)m * N + n] = f32_to_bf16(v);
            break;
          case MODE_GELU: {
            v = 0.5f * v * (1.0f + erff(v * 0.70710678118654752f));
            outB[(size_t)m * N + n] = f32_to_bf16(v);
            break;
          }
          case MODE_QV: {  // [B,NH,S,HD]
            const int b = m >> 9, s = m & 511, h = n >> 6, d = n & 63;
            outB[(((size_t)(b * 16 + h)) * 512 + s) * 64 + d] = f32_to_bf16(v);
            break;
          }
          case MODE_KT: {  // [B,NH,HD,S]
            const int b = m >> 9, s = m & 511, h = n >> 6, d = n & 63;
            outB[(((size_t)(b * 16 + h)) * 64 + d) * 512 + s] = f32_to_bf16(v);
            break;
          }
        }
      }
    }
  }
}

// ---------------------------------------------------------------------------
// Flash-style attention. One workgroup = one (b,h), 128 q-rows; 8 waves each
// own 16 q-rows and stream over 8 key-blocks of 64.
//   scores = WMMA(Q, K^T)/8 + mask   (K stored [B,NH,S,HD]  -> B-tile [s][d])
//   ctx   += WMMA(P, V)              (V stored [B,NH,HD,S]  -> B-tile [d][s])
// ---------------------------------------------------------------------------
__global__ __launch_bounds__(256) void attn_fa(
    const u16* __restrict__ Q, const u16* __restrict__ Kg,
    const u16* __restrict__ Vt, const float* __restrict__ mask,
    u16* __restrict__ CTX) {
  __shared__ __align__(16) u16 Qs[128 * 64];     // 16 KB [q][d]
  __shared__ __align__(16) u16 Ks[64 * 64];      // 8 KB  [s][d] (n-major, k=d)
  __shared__ __align__(16) u16 Vs[64 * 64];      // 8 KB  [d][s] (n-major, k=s)
  __shared__ __align__(16) u16 Ps[8][16 * 64];   // 16 KB per-wave P tile [q][s]

  const int tid = threadIdx.x, lane = tid & 31, wave = tid >> 5;
  const int bh = blockIdx.y, b = bh >> 4, h = bh & 15;
  const int q0 = blockIdx.x * 128;
  const int half = lane >> 4, ln = lane & 15;

  {  // async-stage Q tile [128 x 64]
    const int row = tid >> 1, c0 = (tid & 1) * 32;
    const u16* g = Q + ((size_t)bh * 512 + q0 + row) * 64 + c0;
    u16* l = Qs + row * 64 + c0;
#pragma unroll
    for (int c = 0; c < 4; ++c) stage_async128(l + c * 8, g + c * 8);
  }
  wait_async0();
  __syncthreads();

  const v16bf qa0 = load_frag16(Qs, wave * 16, 0, 64, lane);
  const v16bf qa1 = load_frag16(Qs, wave * 16, 32, 64, lane);

  v8f ctx[4];
#pragma unroll
  for (int df = 0; df < 4; ++df) ctx[df] = zero8();
  float mrun[8], lrun[8];
#pragma unroll
  for (int r = 0; r < 8; ++r) { mrun[r] = -1e30f; lrun[r] = 0.f; }

  for (int j = 0; j < 8; ++j) {
    {  // async-stage K block [64s x 64d] and V block [64d x 64s]
      const int r4 = tid >> 2, c4 = (tid & 3) * 16;
      const u16* gk = Kg + ((size_t)bh * 512 + j * 64 + r4) * 64 + c4;
      const u16* gv = Vt + ((size_t)bh * 64 + r4) * 512 + j * 64 + c4;
      u16* lk = Ks + r4 * 64 + c4;
      u16* lv = Vs + r4 * 64 + c4;
      stage_async128(lk, gk);
      stage_async128(lk + 8, gk + 8);
      stage_async128(lv, gv);
      stage_async128(lv + 8, gv + 8);
    }
    wait_async0();
    __syncthreads();

    // scores: 16 q-rows x 64 keys, K-dim = HD = 64 (two WMMA k-steps)
    v8f sc[4];
#pragma unroll
    for (int nf = 0; nf < 4; ++nf) {
      v8f s = zero8();
      s = wmma_bf16(qa0, load_frag16(Ks, nf * 16, 0, 64, lane), s);
      s = wmma_bf16(qa1, load_frag16(Ks, nf * 16, 32, 64, lane), s);
      const float mk = mask[(size_t)b * 512 + j * 64 + nf * 16 + ln];
#pragma unroll
      for (int r = 0; r < 8; ++r) s[r] = s[r] * 0.125f + mk;  // 1/sqrt(64)
      sc[nf] = s;
    }

    // Online softmax; row r+8*half lives across the 16 lanes of the half.
    float corr[8];
#pragma unroll
    for (int r = 0; r < 8; ++r) {
      float v = fmaxf(fmaxf(sc[0][r], sc[1][r]), fmaxf(sc[2][r], sc[3][r]));
      v = fmaxf(v, __shfl_xor(v, 1));
      v = fmaxf(v, __shfl_xor(v, 2));
      v = fmaxf(v, __shfl_xor(v, 4));
      v = fmaxf(v, __shfl_xor(v, 8));
      const float mn = fmaxf(mrun[r], v);
      corr[r] = __expf(mrun[r] - mn);
      mrun[r] = mn;
    }
#pragma unroll
    for (int nf = 0; nf < 4; ++nf)
#pragma unroll
      for (int r = 0; r < 8; ++r) sc[nf][r] = __expf(sc[nf][r] - mrun[r]);
#pragma unroll
    for (int r = 0; r < 8; ++r) {
      float s = sc[0][r] + sc[1][r] + sc[2][r] + sc[3][r];
      s += __shfl_xor(s, 1);
      s += __shfl_xor(s, 2);
      s += __shfl_xor(s, 4);
      s += __shfl_xor(s, 8);
      lrun[r] = lrun[r] * corr[r] + s;
    }
#pragma unroll
    for (int df = 0; df < 4; ++df)
#pragma unroll
      for (int r = 0; r < 8; ++r) ctx[df][r] *= corr[r];

    // C-frag -> A-frag relayout through per-wave LDS (in-order within a wave).
    u16* P = &Ps[wave][0];
#pragma unroll
    for (int nf = 0; nf < 4; ++nf)
#pragma unroll
      for (int r = 0; r < 8; ++r)
        P[(r + half * 8) * 64 + nf * 16 + ln] = f32_to_bf16(sc[nf][r]);
    const v16bf pa0 = load_frag16(P, 0, 0, 64, lane);
    const v16bf pa1 = load_frag16(P, 0, 32, 64, lane);
#pragma unroll
    for (int df = 0; df < 4; ++df) {
      ctx[df] = wmma_bf16(pa0, load_frag16(Vs, df * 16, 0, 64, lane), ctx[df]);
      ctx[df] = wmma_bf16(pa1, load_frag16(Vs, df * 16, 32, 64, lane), ctx[df]);
    }
    __syncthreads();
  }

  // Normalize and scatter ctx back to [B,S,H] bf16 row-major.
#pragma unroll
  for (int df = 0; df < 4; ++df)
#pragma unroll
    for (int r = 0; r < 8; ++r) {
      const float v = ctx[df][r] / lrun[r];
      const int tok = b * 512 + q0 + wave * 16 + r + half * 8;
      const int col = h * 64 + df * 16 + ln;
      CTX[(size_t)tok * 1024 + col] = f32_to_bf16(v);
    }
}

// ---------------------------------------------------------------------------
// Fused residual + LayerNorm over H=1024. One block per row, 256 threads.
// ---------------------------------------------------------------------------
__global__ __launch_bounds__(256) void res_ln(
    const float* __restrict__ Y, const float* __restrict__ R,
    const float* __restrict__ g, const float* __restrict__ be,
    float* __restrict__ outF, u16* __restrict__ outB) {
  __shared__ float red[16];
  const int row = blockIdx.x, tid = threadIdx.x;
  const float* y = Y + (size_t)row * 1024;
  const float* r = R + (size_t)row * 1024;

  float x[4], s = 0.f, s2 = 0.f;
#pragma unroll
  for (int i = 0; i < 4; ++i) {
    x[i] = y[tid + i * 256] + r[tid + i * 256];
    s += x[i];
    s2 += x[i] * x[i];
  }
#pragma unroll
  for (int o = 1; o < 32; o <<= 1) {
    s += __shfl_xor(s, o);
    s2 += __shfl_xor(s2, o);
  }
  if ((tid & 31) == 0) { red[tid >> 5] = s; red[8 + (tid >> 5)] = s2; }
  __syncthreads();
  float ts = 0.f, ts2 = 0.f;
#pragma unroll
  for (int w = 0; w < 8; ++w) { ts += red[w]; ts2 += red[8 + w]; }
  const float mu = ts * (1.0f / 1024.0f);
  const float var = ts2 * (1.0f / 1024.0f) - mu * mu;
  const float rs = rsqrtf(var + 1e-5f);
#pragma unroll
  for (int i = 0; i < 4; ++i) {
    const int c = tid + i * 256;
    const float o = (x[i] - mu) * rs * g[c] + be[c];
    outF[(size_t)row * 1024 + c] = o;
    if (outB) outB[(size_t)row * 1024 + c] = f32_to_bf16(o);
  }
}

// ---------------------------------------------------------------------------
// Host orchestration
// ---------------------------------------------------------------------------
extern "C" void kernel_launch(void* const* d_in, const int* in_sizes, int n_in,
                              void* d_out, int out_size, void* d_ws,
                              size_t ws_size, hipStream_t stream) {
  (void)in_sizes; (void)n_in; (void)out_size; (void)ws_size;
  const float* hs   = (const float*)d_in[0];
  const float* msk  = (const float*)d_in[1];
  const float* wq   = (const float*)d_in[2];
  const float* bq   = (const float*)d_in[3];
  const float* wk   = (const float*)d_in[4];
  const float* bk   = (const float*)d_in[5];
  const float* wv   = (const float*)d_in[6];
  const float* bv   = (const float*)d_in[7];
  const float* wo   = (const float*)d_in[8];
  const float* bo   = (const float*)d_in[9];
  const float* ln1g = (const float*)d_in[10];
  const float* ln1b = (const float*)d_in[11];
  const float* wi   = (const float*)d_in[12];
  const float* bi   = (const float*)d_in[13];
  const float* wo2  = (const float*)d_in[14];
  const float* bo2  = (const float*)d_in[15];
  const float* ln2g = (const float*)d_in[16];
  const float* ln2b = (const float*)d_in[17];
  float* out = (float*)d_out;

  const int M = 4096, Hd = 1024, FFd = 4096;
  char* ws = (char*)d_ws;
  size_t off = 0;
  auto take = [&](size_t bytes) -> char* {
    char* p = ws + off;
    off += (bytes + 511) & ~(size_t)511;
    return p;
  };
  // All weight copies are bf16 TRANSPOSED ([N][K]) for N-major B tiles.
  u16* Xbf   = (u16*)take((size_t)M * Hd * 2);     // 8 MB
  u16* WqT   = (u16*)take((size_t)Hd * Hd * 2);    // 2 MB
  u16* WkT   = (u16*)take((size_t)Hd * Hd * 2);    // 2 MB
  u16* WvT   = (u16*)take((size_t)Hd * Hd * 2);    // 2 MB
  u16* WoT   = (u16*)take((size_t)Hd * Hd * 2);    // 2 MB
  u16* WiT   = (u16*)take((size_t)Hd * FFd * 2);   // 8 MB
  u16* Wo2T  = (u16*)take((size_t)FFd * Hd * 2);   // 8 MB
  u16* Qb    = (u16*)take((size_t)M * Hd * 2);     // 8 MB (reused by Ib)
  u16* Kb    = (u16*)take((size_t)M * Hd * 2);     // 8 MB (reused by Ib)
  u16* Vtb   = (u16*)take((size_t)M * Hd * 2);     // 8 MB (reused by Ib)
  u16* Ctxb  = (u16*)take((size_t)M * Hd * 2);     // 8 MB (reused by Ib)
  float* Ybuf = (float*)take((size_t)M * Hd * 4);  // 16 MB (reused for Z)
  float* Aln  = (float*)take((size_t)M * Hd * 4);  // 16 MB
  u16* Alnb   = (u16*)take((size_t)M * Hd * 2);    // 8 MB
  u16* Ib = Qb;  // FFN intermediate [M, FF] bf16 reuses Q/K/V/Ctx (32 MB contig)

  const dim3 blk(256);
  cast_bf16<<<4096, blk, 0, stream>>>(hs, Xbf, M * Hd);
  cast_transpose_bf16<<<dim3(Hd / 32, Hd / 32), blk, 0, stream>>>(wq, WqT, Hd, Hd);
  cast_transpose_bf16<<<dim3(Hd / 32, Hd / 32), blk, 0, stream>>>(wk, WkT, Hd, Hd);
  cast_transpose_bf16<<<dim3(Hd / 32, Hd / 32), blk, 0, stream>>>(wv, WvT, Hd, Hd);
  cast_transpose_bf16<<<dim3(Hd / 32, Hd / 32), blk, 0, stream>>>(wo, WoT, Hd, Hd);
  cast_transpose_bf16<<<dim3(FFd / 32, Hd / 32), blk, 0, stream>>>(wi, WiT, Hd, FFd);
  cast_transpose_bf16<<<dim3(Hd / 32, FFd / 32), blk, 0, stream>>>(wo2, Wo2T, FFd, Hd);

  const dim3 gH(Hd / 128, M / 128);   // (8, 32)
  const dim3 gF(FFd / 128, M / 128);  // (32, 32)
  gemm_bf16<<<gH, blk, 0, stream>>>(Xbf, WqT, bq, nullptr, Qb,  M, Hd, Hd, MODE_QV);
  gemm_bf16<<<gH, blk, 0, stream>>>(Xbf, WkT, bk, nullptr, Kb,  M, Hd, Hd, MODE_QV);
  gemm_bf16<<<gH, blk, 0, stream>>>(Xbf, WvT, bv, nullptr, Vtb, M, Hd, Hd, MODE_KT);

  attn_fa<<<dim3(4, 128), blk, 0, stream>>>(Qb, Kb, Vtb, msk, Ctxb);

  gemm_bf16<<<gH, blk, 0, stream>>>(Ctxb, WoT, bo, Ybuf, nullptr, M, Hd, Hd, MODE_F32);
  res_ln<<<4096, blk, 0, stream>>>(Ybuf, hs, ln1g, ln1b, Aln, Alnb);

  gemm_bf16<<<gF, blk, 0, stream>>>(Alnb, WiT, bi, nullptr, Ib, M, FFd, Hd, MODE_GELU);
  gemm_bf16<<<gH, blk, 0, stream>>>(Ib, Wo2T, bo2, Ybuf, nullptr, M, Hd, FFd, MODE_F32);
  res_ln<<<4096, blk, 0, stream>>>(Ybuf, Aln, ln2g, ln2b, out, nullptr);
}
